// SparseMoE_27865747816886
// MI455X (gfx1250) — matrix operand order repaired and checked by reading, not compiled, not run on hardware
//
#include <hip/hip_runtime.h>
#include <stdint.h>

// Problem constants (fixed by the reference)
#define T_TOK 4096   // B*S
#define HDIM  768
#define FFDIM 1536
#define NEXP  8

typedef __attribute__((ext_vector_type(16))) __bf16          v16bf;
typedef __attribute__((ext_vector_type(16))) unsigned short  v16u;
typedef __attribute__((ext_vector_type(8)))  unsigned short  v8u;
typedef __attribute__((ext_vector_type(8)))  float           v8f;

// ---------------------------------------------------------------- helpers ---
__device__ __forceinline__ unsigned short f2bf(float f) {
  union { float f; unsigned int u; } v; v.f = f;
  unsigned int u = v.u;
  // round-to-nearest-even fp32 -> bf16
  return (unsigned short)((u + 0x7FFFu + ((u >> 16) & 1u)) >> 16);
}

__device__ __forceinline__ float gelu_erf(float x) {
  return 0.5f * x * (1.0f + erff(x * 0.70710678118654752440f));
}

__device__ __forceinline__ v16bf make_frag(v8u lo, v8u hi) {
  v16u t = __builtin_shufflevector(lo, hi, 0,1,2,3,4,5,6,7,8,9,10,11,12,13,14,15);
  return __builtin_bit_cast(v16bf, t);
}

// A fragment (16x32 bf16, rows = M): lane L holds row (L&15);
// lanes 0-15 cover K {0..7,16..23}, lanes 16-31 cover K {8..15,24..31}.
__device__ __forceinline__ v16bf load_a_frag(const unsigned short* p) {
  v8u lo = *(const v8u*)(p);
  v8u hi = *(const v8u*)(p + 16);
  return make_frag(lo, hi);
}

// B fragment (32x16 bf16, cols = N): lane L holds column (L&15);
// lanes 0-15 cover K {0..7,16..23}, lanes 16-31 cover K {8..15,24..31}.
__device__ __forceinline__ v16bf load_b_frag(const unsigned short* p) {
  v8u lo = *(const v8u*)(p);
  v8u hi = *(const v8u*)(p + 16);
  return make_frag(lo, hi);
}

#define WMMA_BF16(a, b, c) \
  __builtin_amdgcn_wmma_f32_16x16x32_bf16(false, (a), false, (b), (short)0, (c), false, false)

// CDNA5 async global->LDS copy (ASYNCcnt-tracked DMA into LDS), GV addressing.
__device__ __forceinline__ void async_b128_to_lds(unsigned lds_byte_off, const void* gaddr) {
  asm volatile("global_load_async_to_lds_b128 %0, %1, off"
               :: "v"(lds_byte_off), "v"(gaddr) : "memory");
}
__device__ __forceinline__ void wait_async0() {
  asm volatile("s_wait_asynccnt 0x0" ::: "memory");
}

// ------------------------------------------------------------ gate kernel ---
// One wave per token: logits[t,e] = dot(x[t], gate_w[e]); top-2 softmax -> g.
__global__ void __launch_bounds__(256)
moe_gate_topk_kernel(const float* __restrict__ X,   // [T, H]
                     const float* __restrict__ GW,  // [E, H]
                     float* __restrict__ logits,    // [T, E] (second output)
                     float* __restrict__ G)         // [T, E] combined gates
{
  const int lane  = threadIdx.x & 31;
  const int token = blockIdx.x * 8 + (threadIdx.x >> 5);

  float acc[NEXP];
#pragma unroll
  for (int e = 0; e < NEXP; ++e) acc[e] = 0.0f;

  const float* xp = X + (size_t)token * HDIM;
  for (int k = lane; k < HDIM; k += 32) {
    float xv = xp[k];
#pragma unroll
    for (int e = 0; e < NEXP; ++e) acc[e] += xv * GW[e * HDIM + k];
  }
#pragma unroll
  for (int e = 0; e < NEXP; ++e) {
    float v = acc[e];
#pragma unroll
    for (int off = 16; off > 0; off >>= 1) v += __shfl_down(v, off, 32);
    acc[e] = v;
  }
  if (lane == 0) {
    int   i1 = 0;    float l1 = acc[0];
#pragma unroll
    for (int e = 1; e < NEXP; ++e) if (acc[e] > l1) { l1 = acc[e]; i1 = e; }
    int   i2 = -1;   float l2 = -3.402823466e38f;
#pragma unroll
    for (int e = 0; e < NEXP; ++e) if (e != i1 && acc[e] > l2) { l2 = acc[e]; i2 = e; }
    float e2 = expf(l2 - l1);          // softmax over {l1, l2}, max-shifted
    float s  = 1.0f + e2;
    float g1 = 1.0f / s, g2 = e2 / s;
#pragma unroll
    for (int e = 0; e < NEXP; ++e) {
      logits[token * NEXP + e] = acc[e];
      G[token * NEXP + e] = (e == i1) ? g1 : ((e == i2) ? g2 : 0.0f);
    }
  }
}

// ------------------------------------------------------ fp32 -> bf16 pass ---
__global__ void __launch_bounds__(256)
cvt_f32_bf16_kernel(const float* __restrict__ in, unsigned short* __restrict__ out, int n)
{
  int i      = blockIdx.x * blockDim.x + threadIdx.x;
  int stride = gridDim.x * blockDim.x;
  for (; i < n; i += stride) out[i] = f2bf(in[i]);
}

// --------------------------------------------- GEMM1: hmid = gelu(x @ WfcT) -
// A: x_bf16 [M,K] row-major; W: w_fc[e] [N,K] row-major; Hm: [M,N] bf16.
// Block = 8 waves covering a 128x64 C tile; wave = 16x64 strip (4 WMMA accs).
// B tile (64x32 bf16 = 4KB) is shared by all 8 waves -> staged in LDS via
// async global->LDS DMA, double-buffered so the copy overlaps the WMMAs.
__global__ void __launch_bounds__(256)
moe_fc_gelu_kernel(const unsigned short* __restrict__ A,
                   const unsigned short* __restrict__ W,
                   unsigned short* __restrict__ Hm,
                   int M, int N, int K)
{
  __shared__ alignas(16) unsigned short ldsB[2][64 * 32];

  const int tid  = threadIdx.x;
  const int lane = tid & 31;
  const int wave = tid >> 5;
  const int m0   = blockIdx.x * 128 + wave * 16;
  const int n0   = blockIdx.y * 64;

  // Cooperative staging map: thread t -> B row (t>>2), 16B chunk (t&3).
  const int srow   = tid >> 2;
  const int schunk = tid & 3;
  const unsigned short* gB = W + (size_t)(n0 + srow) * K + schunk * 8;
  const unsigned lbase  = (unsigned)(uintptr_t)(&ldsB[0][0]);   // LDS byte offset
  const unsigned ldst0  = lbase + (unsigned)(srow * 64 + schunk * 16);
  const unsigned lbufsz = 64 * 32 * 2;

  const unsigned short* ap = A + (size_t)(m0 + (lane & 15)) * K + ((lane >> 4) << 3);
  // per-lane B fragment offset within an LDS buffer (elements)
  const int foff = ((lane & 15) << 5) + ((lane >> 4) << 3);

  const int steps = K >> 5;  // K / 32

  // prologue: stage K-step 0 into buffer 0
  async_b128_to_lds(ldst0, gB);
  wait_async0();
  __syncthreads();

  v8f acc0 = {}, acc1 = {}, acc2 = {}, acc3 = {};
  for (int s = 0; s < steps; ++s) {
    const int k0  = s << 5;
    const int buf = s & 1;
    if (s + 1 < steps)   // overlap next tile's DMA with this tile's WMMAs
      async_b128_to_lds(ldst0 + (unsigned)((buf ^ 1) * lbufsz), gB + k0 + 32);

    v16bf a = load_a_frag(ap + k0);
    const unsigned short* bb = &ldsB[buf][0] + foff;
    v16bf b0 = load_b_frag(bb);
    v16bf b1 = load_b_frag(bb + 512);      // +16 cols * 32 elems/row
    v16bf b2 = load_b_frag(bb + 1024);
    v16bf b3 = load_b_frag(bb + 1536);
    acc0 = WMMA_BF16(a, b0, acc0);
    acc1 = WMMA_BF16(a, b1, acc1);
    acc2 = WMMA_BF16(a, b2, acc2);
    acc3 = WMMA_BF16(a, b3, acc3);

    if (s + 1 < steps) { wait_async0(); __syncthreads(); }
  }

  // C/D layout: row = m0 + (lane>>4)*8 + i (vgpr i), col = n0 + (lane&15)
  const int r0 = m0 + ((lane >> 4) << 3);
  const int c0 = n0 + (lane & 15);
#pragma unroll
  for (int i = 0; i < 8; ++i) {
    size_t row = (size_t)(r0 + i) * N + c0;
    Hm[row +  0] = f2bf(gelu_erf(acc0[i]));
    Hm[row + 16] = f2bf(gelu_erf(acc1[i]));
    Hm[row + 32] = f2bf(gelu_erf(acc2[i]));
    Hm[row + 48] = f2bf(gelu_erf(acc3[i]));
  }
}

// ------------------------------- GEMM2: out (+)= g[:,e] * (hmid @ WprojT) ---
__global__ void __launch_bounds__(256)
moe_proj_combine_kernel(const unsigned short* __restrict__ A,   // hmid [M,K]
                        const unsigned short* __restrict__ W,   // w_proj[e] [N,K]
                        const float* __restrict__ G,            // [M, NEXP]
                        float* __restrict__ Out,                // [M, N] fp32
                        int M, int N, int K, int expert, int accum)
{
  __shared__ alignas(16) unsigned short ldsB[2][64 * 32];

  const int tid  = threadIdx.x;
  const int lane = tid & 31;
  const int wave = tid >> 5;
  const int m0   = blockIdx.x * 128 + wave * 16;
  const int n0   = blockIdx.y * 64;

  const int srow   = tid >> 2;
  const int schunk = tid & 3;
  const unsigned short* gB = W + (size_t)(n0 + srow) * K + schunk * 8;
  const unsigned lbase  = (unsigned)(uintptr_t)(&ldsB[0][0]);
  const unsigned ldst0  = lbase + (unsigned)(srow * 64 + schunk * 16);
  const unsigned lbufsz = 64 * 32 * 2;

  const unsigned short* ap = A + (size_t)(m0 + (lane & 15)) * K + ((lane >> 4) << 3);
  const int foff = ((lane & 15) << 5) + ((lane >> 4) << 3);

  const int steps = K >> 5;

  async_b128_to_lds(ldst0, gB);
  wait_async0();
  __syncthreads();

  v8f acc0 = {}, acc1 = {}, acc2 = {}, acc3 = {};
  for (int s = 0; s < steps; ++s) {
    const int k0  = s << 5;
    const int buf = s & 1;
    if (s + 1 < steps)
      async_b128_to_lds(ldst0 + (unsigned)((buf ^ 1) * lbufsz), gB + k0 + 32);

    v16bf a = load_a_frag(ap + k0);
    const unsigned short* bb = &ldsB[buf][0] + foff;
    v16bf b0 = load_b_frag(bb);
    v16bf b1 = load_b_frag(bb + 512);
    v16bf b2 = load_b_frag(bb + 1024);
    v16bf b3 = load_b_frag(bb + 1536);
    acc0 = WMMA_BF16(a, b0, acc0);
    acc1 = WMMA_BF16(a, b1, acc1);
    acc2 = WMMA_BF16(a, b2, acc2);
    acc3 = WMMA_BF16(a, b3, acc3);

    if (s + 1 < steps) { wait_async0(); __syncthreads(); }
  }

  const int r0 = m0 + ((lane >> 4) << 3);
  const int c0 = n0 + (lane & 15);
#pragma unroll
  for (int i = 0; i < 8; ++i) {
    const int   r  = r0 + i;
    const float sg = G[r * NEXP + expert];
    size_t row = (size_t)r * N + c0;
    float p0 = accum ? Out[row +  0] : 0.0f;
    float p1 = accum ? Out[row + 16] : 0.0f;
    float p2 = accum ? Out[row + 32] : 0.0f;
    float p3 = accum ? Out[row + 48] : 0.0f;
    Out[row +  0] = p0 + sg * acc0[i];
    Out[row + 16] = p1 + sg * acc1[i];
    Out[row + 32] = p2 + sg * acc2[i];
    Out[row + 48] = p3 + sg * acc3[i];
  }
}

// ---------------------------------------------------------------- launch ----
extern "C" void kernel_launch(void* const* d_in, const int* in_sizes, int n_in,
                              void* d_out, int out_size, void* d_ws, size_t ws_size,
                              hipStream_t stream) {
  const float* x     = (const float*)d_in[0];  // [B,S,H]
  const float* gw    = (const float*)d_in[1];  // [E,H]
  const float* wfc   = (const float*)d_in[2];  // [E,FF,H]
  const float* wproj = (const float*)d_in[3];  // [E,H,FF]

  float* out    = (float*)d_out;                    // [T,H]
  float* logits = out + (size_t)T_TOK * HDIM;       // [T,E]

  // Workspace carve-out (all chunk sizes are 256B multiples)
  char* ws = (char*)d_ws;
  float*          Gg   = (float*)ws;          ws += (size_t)T_TOK * NEXP * 4;
  unsigned short* xb   = (unsigned short*)ws; ws += (size_t)T_TOK * HDIM * 2;
  unsigned short* wfcb = (unsigned short*)ws; ws += (size_t)NEXP * FFDIM * HDIM * 2;
  unsigned short* wpb  = (unsigned short*)ws; ws += (size_t)NEXP * HDIM * FFDIM * 2;
  unsigned short* hm   = (unsigned short*)ws; ws += (size_t)T_TOK * FFDIM * 2;
  (void)ws_size; (void)in_sizes; (void)n_in; (void)out_size;

  // 1) gating: logits + top-2 softmax combined gates
  moe_gate_topk_kernel<<<T_TOK / 8, 256, 0, stream>>>(x, gw, logits, Gg);

  // 2) one-time bf16 conversions (halves all GEMM read traffic)
  cvt_f32_bf16_kernel<<<2048, 256, 0, stream>>>(x,     xb,   T_TOK * HDIM);
  cvt_f32_bf16_kernel<<<4096, 256, 0, stream>>>(wfc,   wfcb, NEXP * FFDIM * HDIM);
  cvt_f32_bf16_kernel<<<4096, 256, 0, stream>>>(wproj, wpb,  NEXP * HDIM * FFDIM);

  // 3) dense-over-experts (gates are zero for non-selected experts, so this
  //    matches the reference math exactly); same-stream ordering makes the
  //    read-modify-write accumulation into `out` safe.
  for (int e = 0; e < NEXP; ++e) {
    moe_fc_gelu_kernel<<<dim3(T_TOK / 128, FFDIM / 64), dim3(256), 0, stream>>>(
        xb, wfcb + (size_t)e * FFDIM * HDIM, hm, T_TOK, FFDIM, HDIM);
    moe_proj_combine_kernel<<<dim3(T_TOK / 128, HDIM / 64), dim3(256), 0, stream>>>(
        hm, wpb + (size_t)e * HDIM * FFDIM, Gg, out, T_TOK, HDIM, FFDIM, e, e != 0);
  }
}
